// MMDoubleStreamBlock_80822694576629
// MI455X (gfx1250) — matrix operand order repaired and checked by reading, not compiled, not run on hardware
//
#include <hip/hip_runtime.h>
#include <cstdint>

// ---------------------------------------------------------------------------
// MMDiT double-stream block for MI455X (gfx1250, wave32, WMMA).
//   H=2048, NH=16, HD=128, MLP=8192, S_img=2048, S_txt=256, S=2304, fp32 I/O.
// All dense math goes through v_wmma_f32_16x16x32_f16 (f16 in, f32 accum).
// Weights stay fp32 in HBM and are converted to f16 while staged into
// ping-pong LDS (one tile shared by 8 waves, BM=256 -> low L2 re-read).
// Activations are cast to f16 once per stage (L2 resident at 192MB).
// Workspace use (~113 MB, assumed <= ws_size):
//   sv(8KB) | mod_img(48KB) | mod_txt(48KB) | qkv f32 (56.6MB, aliased by h1)
//   | xin f16 (9.4MB, aliased by attn out) | qkvh f16 (28.3MB) | res f32 (18.9MB)
// ---------------------------------------------------------------------------

#define HDIM   2048
#define NHEAD  16
#define HEADD  128
#define MLPD   8192
#define SIMG   2048
#define STXT   256
#define SEQ    2304
#define EPSV   1e-6f

typedef _Float16 f16;
typedef __attribute__((ext_vector_type(16))) _Float16 v16h;
typedef __attribute__((ext_vector_type(8)))  _Float16 v8h;
typedef __attribute__((ext_vector_type(4)))  _Float16 v4h;
typedef __attribute__((ext_vector_type(8)))  float    v8f;
typedef __attribute__((ext_vector_type(4)))  float    v4f;

__device__ __forceinline__ v8f vzero8() {
  v8f z;
#pragma unroll
  for (int i = 0; i < 8; ++i) z[i] = 0.f;
  return z;
}

// A-fragment (16x32 f16, ISA 7.12.2 interleaved K-groups):
// lane r=lane&15 -> row r ; half h=lane>>4 ; elements e<8 -> K=kb+8h+e,
// e>=8 -> K=kb+16+8h+(e-8).  Two 16B loads.
__device__ __forceinline__ v16h load_afrag(const f16* p, int ld, int kb) {
  int lane = threadIdx.x & 31;
  int r = lane & 15, h = (lane >> 4) & 1;
  const f16* q = p + (size_t)r * ld + kb + 8 * h;
  v8h lo = *(const v8h*)q;
  v8h hi = *(const v8h*)(q + 16);
  v16h o;
#pragma unroll
  for (int i = 0; i < 8; ++i) { o[i] = lo[i]; o[8 + i] = hi[i]; }
  return o;
}

// B-fragment (32x16 f16): lane r = column, half h ; elements e<8 -> K=kb+8h+e,
// e>=8 -> K=kb+16+8h+(e-8) (same order as A). Source rows are
// columns of B (i.e., B^T row-major, K contiguous). Two 16B loads.
__device__ __forceinline__ v16h load_bfrag(const f16* p, int ld, int kb) {
  int lane = threadIdx.x & 31;
  int r = lane & 15, h = (lane >> 4) & 1;
  const f16* q = p + (size_t)r * ld + kb + 8 * h;
  v8h lo = *(const v8h*)q;
  v8h hi = *(const v8h*)(q + 16);
  v16h o;
#pragma unroll
  for (int i = 0; i < 8; ++i) { o[i] = lo[i]; o[8 + i] = hi[i]; }
  return o;
}

__device__ __forceinline__ v8f wmma_f16(v16h a, v16h b, v8f c) {
  return __builtin_amdgcn_wmma_f32_16x16x32_f16(false, a, false, b, (short)0,
                                                c, false, false);
}

__device__ __forceinline__ v8h cvt8h(const float* p) {
  v4f a0 = *(const v4f*)p;
  v4f a1 = *(const v4f*)(p + 4);
  v8h h;
#pragma unroll
  for (int i = 0; i < 4; ++i) { h[i] = (f16)a0[i]; h[4 + i] = (f16)a1[i]; }
  return h;
}

__device__ __forceinline__ float gelu_tanh(float x) {
  float x3 = x * x * x;
  return 0.5f * x * (1.f + tanhf(0.7978845608028654f * (x + 0.044715f * x3)));
}

// ---------------------------------------------------------------------------
// 1) sv = silu(vec)
__global__ __launch_bounds__(256) void silu_kernel(const float* __restrict__ v,
                                                   float* __restrict__ sv) {
  int i = blockIdx.x * 256 + threadIdx.x;
  float x = v[i];
  sv[i] = x / (1.f + __expf(-x));
}

// 2) mod = sv @ W^T + b   (one wave per output, 6H outputs; b128 streaming)
__global__ __launch_bounds__(256) void mod_kernel(const float* __restrict__ sv,
                                                  const float* __restrict__ w,
                                                  const float* __restrict__ b,
                                                  float* __restrict__ out) {
  int wid = threadIdx.x >> 5, lane = threadIdx.x & 31;
  int o = blockIdx.x * 8 + wid;
  const float* wr = w + (size_t)o * HDIM;
  float s = 0.f;
#pragma unroll 4
  for (int i = 0; i < HDIM / 128; ++i) {
    int k = lane * 4 + 128 * i;
    v4f a = *(const v4f*)(sv + k);
    v4f c = *(const v4f*)(wr + k);
    s += a[0] * c[0] + a[1] * c[1] + a[2] * c[2] + a[3] * c[3];
  }
#pragma unroll
  for (int off = 16; off; off >>= 1) s += __shfl_xor(s, off);
  if (lane == 0) out[o] = s + b[o];
}

// 3) xin = LN(x)*(1+sc) + sh   -> f16   (one block per row, b128 loads)
__global__ __launch_bounds__(256) void lnmod_kernel(
    const float* __restrict__ pImg, const float* __restrict__ pTxt,
    const float* __restrict__ modImg, const float* __restrict__ modTxt,
    int shBase, f16* __restrict__ out) {
  __shared__ float red[18];
  int r = blockIdx.x, tid = threadIdx.x;
  const float* x   = (r < SIMG) ? pImg + (size_t)r * HDIM
                                : pTxt + (size_t)(r - SIMG) * HDIM;
  const float* mod = (r < SIMG) ? modImg : modTxt;
  v4f xv[2];
  float s1 = 0.f, s2 = 0.f;
#pragma unroll
  for (int i = 0; i < 2; ++i) {
    xv[i] = *(const v4f*)(x + tid * 4 + 1024 * i);
#pragma unroll
    for (int e = 0; e < 4; ++e) { s1 += xv[i][e]; s2 += xv[i][e] * xv[i][e]; }
  }
#pragma unroll
  for (int off = 16; off; off >>= 1) { s1 += __shfl_xor(s1, off); s2 += __shfl_xor(s2, off); }
  int wid = tid >> 5;
  if ((tid & 31) == 0) { red[wid] = s1; red[8 + wid] = s2; }
  __syncthreads();
  if (tid == 0) {
    float a = 0.f, b = 0.f;
#pragma unroll
    for (int i = 0; i < 8; ++i) { a += red[i]; b += red[8 + i]; }
    float m = a * (1.f / HDIM);
    float var = b * (1.f / HDIM) - m * m;
    red[16] = m;
    red[17] = rsqrtf(var + EPSV);
  }
  __syncthreads();
  float m = red[16], rs = red[17];
#pragma unroll
  for (int i = 0; i < 2; ++i) {
    int c = tid * 4 + 1024 * i;
    v4f sh = *(const v4f*)(mod + shBase + c);
    v4f sc = *(const v4f*)(mod + shBase + HDIM + c);
    v4h y;
#pragma unroll
    for (int e = 0; e < 4; ++e)
      y[e] = (f16)((xv[i][e] - m) * rs * (1.f + sc[e]) + sh[e]);
    *(v4h*)(out + (size_t)r * HDIM + c) = y;
  }
}

// 4) WMMA GEMM: C[M,N] = A_f16[M,K] @ W_f32[N,K]^T + bias, fused epilogues.
//    Block: 256 thr (8 waves), tile 256M x 64N, K-step 32.
//    W converted fp32->f16 into ping-pong LDS; one barrier per K-step; next
//    W tile is loaded to registers before the barrier (latency hiding).
//    EPI 0: Cf = c (f32)   EPI 1: Ch = gelu(c) (f16)
//    EPI 2: Cf = X + G[col]*c (residual + per-column gate)
template <int EPI>
__global__ __launch_bounds__(256) void gemm_kernel(
    const f16* __restrict__ A, int lda, const float* __restrict__ W, int ldw,
    int K, const float* __restrict__ bias, float* __restrict__ Cf,
    f16* __restrict__ Ch, int ldc, const float* __restrict__ X,
    const float* __restrict__ G) {
  __shared__ f16 Wt[2][64 * 32];
  int wv = threadIdx.x >> 5, lane = threadIdx.x & 31;
  int row0 = blockIdx.x * 256 + 16 * wv;    // wave owns rows row0 & row0+128
  int n0 = blockIdx.y * 64;
  const f16* Ar0 = A + (size_t)row0 * lda;
  const f16* Ar1 = Ar0 + (size_t)128 * lda;

  int tn = threadIdx.x >> 2;          // 0..63 : W row (output column)
  int tk = (threadIdx.x & 3) * 8;     // k offset within 32-tile
  const float* Wp = W + (size_t)(n0 + tn) * ldw + tk;

  v8f acc[2][4];
#pragma unroll
  for (int g = 0; g < 2; ++g)
#pragma unroll
    for (int j = 0; j < 4; ++j) acc[g][j] = vzero8();

  v8h hw = cvt8h(Wp);                 // first tile in registers
  for (int k = 0; k < K; k += 32) {
    int pb = (k >> 5) & 1;
    *(v8h*)&Wt[pb][tn * 32 + tk] = hw;
    if (k + 32 < K) {                 // fetch next tile before the barrier
      hw = cvt8h(Wp + k + 32);
      __builtin_prefetch(Wp + k + 288, 0, 0);   // global_prefetch_b8 ahead
    }
    __syncthreads();
    v16h af0 = load_afrag(Ar0, lda, k);
    v16h af1 = load_afrag(Ar1, lda, k);
#pragma unroll
    for (int j = 0; j < 4; ++j) {
      v16h bf = load_bfrag(&Wt[pb][(16 * j) * 32], 32, 0);
      acc[0][j] = wmma_f16(af0, bf, acc[0][j]);
      acc[1][j] = wmma_f16(af1, bf, acc[1][j]);
    }
  }

  int half = lane >> 4, r = lane & 15;
#pragma unroll
  for (int j = 0; j < 4; ++j) {
    int col = n0 + 16 * j + r;
    float bc = bias[col];
    float g = (EPI == 2) ? G[col] : 0.f;
#pragma unroll
    for (int grp = 0; grp < 2; ++grp) {
#pragma unroll
      for (int v = 0; v < 8; ++v) {
        int row = row0 + 128 * grp + v + 8 * half;
        float c = acc[grp][j][v] + bc;
        if (EPI == 0) Cf[(size_t)row * ldc + col] = c;
        if (EPI == 1) Ch[(size_t)row * ldc + col] = (f16)gelu_tanh(c);
        if (EPI == 2) Cf[(size_t)row * ldc + col] = X[(size_t)row * ldc + col] + g * c;
      }
    }
  }
}

// 5) RMS-norm q/k, RoPE (img rows only), pack q/k/v -> [3][NH][S][HD] f16.
//    One wave per (row, which, head): S*48 waves.
__global__ __launch_bounds__(256) void qkvpost_kernel(
    const float* __restrict__ qkv, const float* __restrict__ cosp,
    const float* __restrict__ sinp, const float* __restrict__ qni,
    const float* __restrict__ kni, const float* __restrict__ qnt,
    const float* __restrict__ knt, f16* __restrict__ qh) {
  int g = blockIdx.x * 8 + (threadIdx.x >> 5);
  int lane = threadIdx.x & 31;
  int row = g / 48, rem = g % 48;
  int which = rem >> 4, h = rem & 15;
  int d0 = lane * 4;
  const float* src = qkv + (size_t)row * (3 * HDIM) + which * HDIM + h * HEADD + d0;
  v4f x = *(const v4f*)src;
  bool isimg = row < SIMG;
  if (which < 2) {
    float ss = x[0] * x[0] + x[1] * x[1] + x[2] * x[2] + x[3] * x[3];
#pragma unroll
    for (int off = 16; off; off >>= 1) ss += __shfl_xor(ss, off);
    float sc = rsqrtf(ss * (1.f / HEADD) + EPSV);
    const float* nw = (which == 0) ? (isimg ? qni : qnt) : (isimg ? kni : knt);
#pragma unroll
    for (int i = 0; i < 4; ++i) x[i] = x[i] * sc * nw[d0 + i];
    if (isimg) {  // RoPE: pairs (d0,d0+1),(d0+2,d0+3)
      int p0 = d0 >> 1;
      float c0 = cosp[row * 64 + p0],     s0 = sinp[row * 64 + p0];
      float c1 = cosp[row * 64 + p0 + 1], s1 = sinp[row * 64 + p0 + 1];
      float e0 = x[0] * c0 - x[1] * s0, o0 = x[1] * c0 + x[0] * s0;
      float e1 = x[2] * c1 - x[3] * s1, o1 = x[3] * c1 + x[2] * s1;
      x[0] = e0; x[1] = o0; x[2] = e1; x[3] = o1;
    }
  }
  v4h xh;
#pragma unroll
  for (int i = 0; i < 4; ++i) xh[i] = (f16)x[i];
  f16* dst = qh + (size_t)which * NHEAD * SEQ * HEADD +
             (size_t)h * SEQ * HEADD + (size_t)row * HEADD + d0;
  *(v4h*)dst = xh;
}

// 6) Flash attention, one head per blockIdx.y, 64 q-rows per blockIdx.x.
//    Block 128 thr (4 waves); wave -> 16 q rows.  QK^T and PV via WMMA.
__global__ __launch_bounds__(128) void attn_kernel(const f16* __restrict__ qh,
                                                   f16* __restrict__ attn_h) {
  __shared__ f16 Vt[HEADD * 64];     // transposed V tile: Vt[d][kv]
  __shared__ f16 Pt[4][16 * 64];     // per-wave probability tile
  const size_t NSH = (size_t)NHEAD * SEQ * HEADD;
  int head = blockIdx.y;
  int qbase = blockIdx.x * 64;
  int wv = threadIdx.x >> 5, lane = threadIdx.x & 31;
  int half = lane >> 4, r = lane & 15;

  const f16* qp = qh + (size_t)head * SEQ * HEADD + (size_t)(qbase + 16 * wv) * HEADD;
  const f16* kp = qh + NSH + (size_t)head * SEQ * HEADD;
  const f16* vp = qh + 2 * NSH + (size_t)head * SEQ * HEADD;

  v16h qa[4];
#pragma unroll
  for (int kk = 0; kk < 4; ++kk) qa[kk] = load_afrag(qp, HEADD, 32 * kk);

  v8f o[8];
  float m[8], l[8];
#pragma unroll
  for (int f = 0; f < 8; ++f) o[f] = vzero8();
#pragma unroll
  for (int v = 0; v < 8; ++v) { m[v] = -1e30f; l[v] = 0.f; }

  const float scale = 0.08838834764831845f;  // 1/sqrt(128)

  for (int kv0 = 0; kv0 < SEQ; kv0 += 64) {
    __syncthreads();
    {  // stage V tile transposed: Vt[d*64 + kv]
      int kvr = threadIdx.x >> 1;
      int dbase = (threadIdx.x & 1) * 64;
      const f16* vrow = vp + (size_t)(kv0 + kvr) * HEADD + dbase;
#pragma unroll
      for (int c = 0; c < 8; ++c) {
        v8h chunk = *(const v8h*)(vrow + 8 * c);
#pragma unroll
        for (int e = 0; e < 8; ++e) Vt[(dbase + 8 * c + e) * 64 + kvr] = chunk[e];
      }
    }
    __syncthreads();

    v8f sc[4];
#pragma unroll
    for (int j = 0; j < 4; ++j) sc[j] = vzero8();
#pragma unroll
    for (int j = 0; j < 4; ++j)
#pragma unroll
      for (int kk = 0; kk < 4; ++kk) {
        v16h bf = load_bfrag(kp + (size_t)(kv0 + 16 * j) * HEADD, HEADD, 32 * kk);
        sc[j] = wmma_f16(qa[kk], bf, sc[j]);
      }

    float p[4][8], tmax[8];
#pragma unroll
    for (int v = 0; v < 8; ++v) tmax[v] = -1e30f;
#pragma unroll
    for (int j = 0; j < 4; ++j)
#pragma unroll
      for (int v = 0; v < 8; ++v) {
        float s = sc[j][v] * scale;
        p[j][v] = s;
        tmax[v] = fmaxf(tmax[v], s);
      }
#pragma unroll
    for (int v = 0; v < 8; ++v)
#pragma unroll
      for (int off = 8; off; off >>= 1) tmax[v] = fmaxf(tmax[v], __shfl_xor(tmax[v], off));

    float alpha[8], rs[8];
#pragma unroll
    for (int v = 0; v < 8; ++v) {
      float nm = fmaxf(m[v], tmax[v]);
      alpha[v] = __expf(m[v] - nm);
      m[v] = nm;
      rs[v] = 0.f;
    }
#pragma unroll
    for (int j = 0; j < 4; ++j)
#pragma unroll
      for (int v = 0; v < 8; ++v) {
        p[j][v] = __expf(p[j][v] - m[v]);
        rs[v] += p[j][v];
      }
#pragma unroll
    for (int v = 0; v < 8; ++v) {
#pragma unroll
      for (int off = 8; off; off >>= 1) rs[v] += __shfl_xor(rs[v], off);
      l[v] = l[v] * alpha[v] + rs[v];
    }
#pragma unroll
    for (int f = 0; f < 8; ++f)
#pragma unroll
      for (int v = 0; v < 8; ++v) o[f][v] *= alpha[v];

    // P -> LDS (C-layout to row-major 16x64), then reload as A fragments.
#pragma unroll
    for (int j = 0; j < 4; ++j)
#pragma unroll
      for (int v = 0; v < 8; ++v)
        Pt[wv][(v + 8 * half) * 64 + 16 * j + r] = (f16)p[j][v];

#pragma unroll
    for (int kk = 0; kk < 2; ++kk) {
      v16h aP = load_afrag(&Pt[wv][0], 64, 32 * kk);
#pragma unroll
      for (int f = 0; f < 8; ++f) {
        v16h bv = load_bfrag(&Vt[(16 * f) * 64], 64, 32 * kk);
        o[f] = wmma_f16(aP, bv, o[f]);
      }
    }
  }

#pragma unroll
  for (int f = 0; f < 8; ++f) {
    int col = head * HEADD + 16 * f + r;
#pragma unroll
    for (int v = 0; v < 8; ++v) {
      int row = qbase + 16 * wv + v + 8 * half;
      attn_h[(size_t)row * HDIM + col] = (f16)(o[f][v] / l[v]);
    }
  }
}

// ---------------------------------------------------------------------------
extern "C" void kernel_launch(void* const* d_in, const int* in_sizes, int n_in,
                              void* d_out, int out_size, void* d_ws,
                              size_t ws_size, hipStream_t stream) {
  (void)in_sizes; (void)n_in; (void)out_size; (void)ws_size;
  const float* img   = (const float*)d_in[0];
  const float* txt   = (const float*)d_in[1];
  const float* vec   = (const float*)d_in[2];
  const float* cosp  = (const float*)d_in[3];
  const float* sinp  = (const float*)d_in[4];
  const float* i_mod_w  = (const float*)d_in[5];
  const float* i_mod_b  = (const float*)d_in[6];
  const float* i_qkv_w  = (const float*)d_in[7];
  const float* i_qkv_b  = (const float*)d_in[8];
  const float* i_qn     = (const float*)d_in[9];
  const float* i_kn     = (const float*)d_in[10];
  const float* i_proj_w = (const float*)d_in[11];
  const float* i_proj_b = (const float*)d_in[12];
  const float* i_fc1_w  = (const float*)d_in[13];
  const float* i_fc1_b  = (const float*)d_in[14];
  const float* i_fc2_w  = (const float*)d_in[15];
  const float* i_fc2_b  = (const float*)d_in[16];
  const float* t_mod_w  = (const float*)d_in[17];
  const float* t_mod_b  = (const float*)d_in[18];
  const float* t_qkv_w  = (const float*)d_in[19];
  const float* t_qkv_b  = (const float*)d_in[20];
  const float* t_qn     = (const float*)d_in[21];
  const float* t_kn     = (const float*)d_in[22];
  const float* t_proj_w = (const float*)d_in[23];
  const float* t_proj_b = (const float*)d_in[24];
  const float* t_fc1_w  = (const float*)d_in[25];
  const float* t_fc1_b  = (const float*)d_in[26];
  const float* t_fc2_w  = (const float*)d_in[27];
  const float* t_fc2_b  = (const float*)d_in[28];
  float* outp = (float*)d_out;

  // workspace carve-up (see header comment)
  float* sv   = (float*)d_ws;
  float* modi = sv + HDIM;
  float* modt = modi + 6 * HDIM;
  float* qkv  = modt + 6 * HDIM;
  f16* xin    = (f16*)(qkv + (size_t)SEQ * 3 * HDIM);
  f16* qh     = xin + (size_t)SEQ * HDIM;
  float* res  = (float*)(qh + (size_t)3 * NHEAD * SEQ * HEADD);
  f16* h1     = (f16*)qkv;   // alias: qkv consumed before fc1 writes h1
  f16* attn_h = xin;         // alias: xin consumed before attention writes

  silu_kernel<<<HDIM / 256, 256, 0, stream>>>(vec, sv);
  mod_kernel<<<(6 * HDIM) / 8, 256, 0, stream>>>(sv, i_mod_w, i_mod_b, modi);
  mod_kernel<<<(6 * HDIM) / 8, 256, 0, stream>>>(sv, t_mod_w, t_mod_b, modt);

  lnmod_kernel<<<SEQ, 256, 0, stream>>>(img, txt, modi, modt, 0, xin);

  gemm_kernel<0><<<dim3(SIMG / 256, (3 * HDIM) / 64), 256, 0, stream>>>(
      xin, HDIM, i_qkv_w, HDIM, HDIM, i_qkv_b, qkv, nullptr, 3 * HDIM, nullptr, nullptr);
  gemm_kernel<0><<<dim3(STXT / 256, (3 * HDIM) / 64), 256, 0, stream>>>(
      xin + (size_t)SIMG * HDIM, HDIM, t_qkv_w, HDIM, HDIM, t_qkv_b,
      qkv + (size_t)SIMG * 3 * HDIM, nullptr, 3 * HDIM, nullptr, nullptr);

  qkvpost_kernel<<<(SEQ * 48) / 8, 256, 0, stream>>>(qkv, cosp, sinp, i_qn,
                                                     i_kn, t_qn, t_kn, qh);

  attn_kernel<<<dim3(SEQ / 64, NHEAD), 128, 0, stream>>>(qh, attn_h);

  gemm_kernel<2><<<dim3(SIMG / 256, HDIM / 64), 256, 0, stream>>>(
      attn_h, HDIM, i_proj_w, HDIM, HDIM, i_proj_b, res, nullptr, HDIM, img,
      modi + 2 * HDIM);
  gemm_kernel<2><<<dim3(STXT / 256, HDIM / 64), 256, 0, stream>>>(
      attn_h + (size_t)SIMG * HDIM, HDIM, t_proj_w, HDIM, HDIM, t_proj_b,
      res + (size_t)SIMG * HDIM, nullptr, HDIM, txt, modt + 2 * HDIM);

  lnmod_kernel<<<SEQ, 256, 0, stream>>>(res, res + (size_t)SIMG * HDIM, modi,
                                        modt, 3 * HDIM, xin);

  gemm_kernel<1><<<dim3(SIMG / 256, MLPD / 64), 256, 0, stream>>>(
      xin, HDIM, i_fc1_w, HDIM, HDIM, i_fc1_b, nullptr, h1, MLPD, nullptr, nullptr);
  gemm_kernel<1><<<dim3(STXT / 256, MLPD / 64), 256, 0, stream>>>(
      xin + (size_t)SIMG * HDIM, HDIM, t_fc1_w, HDIM, HDIM, t_fc1_b, nullptr,
      h1 + (size_t)SIMG * MLPD, MLPD, nullptr, nullptr);

  gemm_kernel<2><<<dim3(SIMG / 256, HDIM / 64), 256, 0, stream>>>(
      h1, MLPD, i_fc2_w, MLPD, MLPD, i_fc2_b, outp, nullptr, HDIM, res,
      modi + 5 * HDIM);
  gemm_kernel<2><<<dim3(STXT / 256, HDIM / 64), 256, 0, stream>>>(
      h1 + (size_t)SIMG * MLPD, MLPD, t_fc2_w, MLPD, MLPD, t_fc2_b,
      outp + (size_t)SIMG * HDIM, nullptr, HDIM, res + (size_t)SIMG * HDIM,
      modt + 5 * HDIM);
}
